// MyModule_5763846111801
// MI455X (gfx1250) — matrix operand order repaired and checked by reading, not compiled, not run on hardware
//
#include <hip/hip_runtime.h>
#include <math.h>

// ---------------- model dims ----------------
#define BB   4
#define TT   512
#define CC   1080
#define HH   20
#define LL   12
#define VV   50257
#define HD   54
#define BT   (BB*TT)     // 2048 rows
#define C3   (3*CC)      // 3240
#define C4   (4*CC)      // 4320
// padded dims (multiples of 32 for K, 16 for N)
#define CP   1088
#define C3P  3264
#define C4P  4352
#define HDP  64
#define BHN  (BB*HH)     // 80
#define VP   50272

typedef __attribute__((ext_vector_type(16))) _Float16 v16h;
typedef __attribute__((ext_vector_type(8)))  _Float16 v8h;
typedef __attribute__((ext_vector_type(8)))  float    v8f;

// ---------------- elementwise helpers ----------------
__device__ __forceinline__ float geluf(float x) {
    const float kk = 0.7978845608028654f; // sqrt(2/pi)
    return 0.5f * x * (1.0f + tanhf(kk * (x + 0.044715f * x * x * x)));
}

// ---------------- embedding: X[b,t,c] = wte[idx]+wpe[t], zero pad ----------------
__global__ void k_embed(const int* __restrict__ idx, const float* __restrict__ wte,
                        const float* __restrict__ wpe, float* __restrict__ X) {
    long long i = (long long)blockIdx.x * 256 + threadIdx.x;
    if (i >= (long long)BT * CP) return;
    int c = (int)(i % CP);
    int row = (int)(i / CP);
    float v = 0.f;
    if (c < CC) {
        int t  = row % TT;
        int id = idx[row];
        v = wte[(long long)id * CC + c] + wpe[(long long)t * CC + c];
    }
    X[i] = v;
}

// ---------------- layernorm (fp32 in, fp16 padded out) ----------------
__global__ void k_ln(const float* __restrict__ X, const float* __restrict__ w,
                     const float* __restrict__ b, _Float16* __restrict__ out) {
    int row = blockIdx.x;
    const float* x = X + (long long)row * CP;
    __shared__ float s1[256], s2[256];
    float sum = 0.f, sq = 0.f;
    for (int c = threadIdx.x; c < CC; c += 256) { float v = x[c]; sum += v; sq += v * v; }
    s1[threadIdx.x] = sum; s2[threadIdx.x] = sq; __syncthreads();
    for (int o = 128; o > 0; o >>= 1) {
        if (threadIdx.x < o) { s1[threadIdx.x] += s1[threadIdx.x + o]; s2[threadIdx.x] += s2[threadIdx.x + o]; }
        __syncthreads();
    }
    float mu  = s1[0] / CC;
    float var = s2[0] / CC - mu * mu;
    float rs  = rsqrtf(var + 1e-5f);
    _Float16* o = out + (long long)row * CP;
    for (int c = threadIdx.x; c < CP; c += 256) {
        float v = (c < CC) ? ((x[c] - mu) * rs * w[c] + b[c]) : 0.f;
        o[c] = (_Float16)v;
    }
}

// ---------------- fp32 [Kv][Nv] -> fp16 [Kp][Np] zero-padded ----------------
__global__ void k_cvt_pad(const float* __restrict__ W, _Float16* __restrict__ Wh,
                          int Kv, int Nv, int Np, long long total) {
    long long i = (long long)blockIdx.x * 256 + threadIdx.x;
    if (i >= total) return;
    int n = (int)(i % Np);
    int k = (int)(i / Np);
    Wh[i] = (_Float16)((k < Kv && n < Nv) ? W[(long long)k * Nv + n] : 0.f);
}

// ---------------- wte [V][C] -> transposed fp16 [CP][VP] for lm-head ----------------
__global__ void k_wteT(const float* __restrict__ wte, _Float16* __restrict__ WT) {
    long long i = (long long)blockIdx.x * 256 + threadIdx.x;
    if (i >= (long long)CP * VP) return;
    int v = (int)(i % VP);
    int c = (int)(i / VP);
    WT[i] = (_Float16)((c < CC && v < VV) ? wte[(long long)v * CC + c] : 0.f);
}

// ---------------- WMMA GEMM: one wave -> 64x16 tile (4 M-subtiles share B) ----------
// A fp16 row-major [M][lda]; B fp16 row-major [K][ldb]; C fp32 [M][ldc]
// flags: bit0 = GELU, bit1 = residual add into C.  Batched via grid.z with strides.
__global__ void k_gemm(const _Float16* __restrict__ A, long long sA, int lda,
                       const _Float16* __restrict__ Bm, long long sB, int ldb,
                       const float* __restrict__ bias,
                       float* __restrict__ C, long long sC, int ldc,
                       _Float16* __restrict__ Ch, long long sCh, int ldch,
                       int K, int Nvalid, int Nstore, int flags) {
    int lane = threadIdx.x;            // blockDim.x == 32 (wave32)
    int z    = blockIdx.z;
    int n0   = blockIdx.x * 16;
    int m0   = blockIdx.y * 64;
    int l15  = lane & 15;
    int hs   = lane >> 4;              // lane-group selector

    const _Float16* Ar0 = A + z * sA + (long long)(m0 + l15) * lda;
    const _Float16* Bb  = Bm + z * sB + n0;
    const long long strideM16 = (long long)16 * lda;

    v8f acc0 = {}, acc1 = {}, acc2 = {}, acc3 = {};
    for (int k0 = 0; k0 < K; k0 += 32) {
        // B fragment: lane = N column (n0+l15); elements 0..7 -> K rows k0+hs*8.., 8..15 -> +16
        v16h b;
        const _Float16* bp = Bb + l15 + (long long)(k0 + hs * 8) * ldb;
        for (int s = 0; s < 8; ++s) { b[s] = bp[(long long)s * ldb]; b[8 + s] = bp[(long long)(16 + s) * ldb]; }
        // A fragments: lane = M row; elements 0..7 -> k0+koff.., 8..15 -> +16
        int ka = k0 + hs * 8;
        v16h a0, a1, a2, a3;
        const _Float16* ap = Ar0 + ka;
        ((v8h*)&a0)[0] = *(const v8h*)(ap);
        ((v8h*)&a0)[1] = *(const v8h*)(ap + 16);
        ap += strideM16;
        ((v8h*)&a1)[0] = *(const v8h*)(ap);
        ((v8h*)&a1)[1] = *(const v8h*)(ap + 16);
        ap += strideM16;
        ((v8h*)&a2)[0] = *(const v8h*)(ap);
        ((v8h*)&a2)[1] = *(const v8h*)(ap + 16);
        ap += strideM16;
        ((v8h*)&a3)[0] = *(const v8h*)(ap);
        ((v8h*)&a3)[1] = *(const v8h*)(ap + 16);

        acc0 = __builtin_amdgcn_wmma_f32_16x16x32_f16(false, a0, false, b, (short)0, acc0, false, false);
        acc1 = __builtin_amdgcn_wmma_f32_16x16x32_f16(false, a1, false, b, (short)0, acc1, false, false);
        acc2 = __builtin_amdgcn_wmma_f32_16x16x32_f16(false, a2, false, b, (short)0, acc2, false, false);
        acc3 = __builtin_amdgcn_wmma_f32_16x16x32_f16(false, a3, false, b, (short)0, acc3, false, false);
    }

    int  n    = n0 + l15;
    bool inN  = (n < Nvalid);
    bool st   = (n < Nstore);
    float bv  = (bias && inN) ? bias[n] : 0.f;
    v8f accs[4] = {acc0, acc1, acc2, acc3};
#pragma unroll
    for (int j = 0; j < 4; ++j) {
#pragma unroll
        for (int r = 0; r < 8; ++r) {
            int   m = m0 + j * 16 + r + hs * 8;  // C/D layout: VGPR r, lanes16-31 -> M=r+8
            float v = accs[j][r] + bv;
            if (flags & 1) v = geluf(v);
            float ov = inN ? v : 0.f;
            if (st) {
                if (C) {
                    float* cp = C + z * sC + (long long)m * ldc + n;
                    *cp = (flags & 2) ? (*cp + ov) : ov;
                }
                if (Ch) Ch[z * sCh + (long long)m * ldch + n] = (_Float16)ov;
            }
        }
    }
}

// ---------------- split QKV -> Qh (scaled), Kt (d-major), Vh ----------------
__global__ void k_split(const float* __restrict__ QKV, _Float16* __restrict__ Qh,
                        _Float16* __restrict__ Kt, _Float16* __restrict__ Vh) {
    long long i = (long long)blockIdx.x * 256 + threadIdx.x;
    if (i >= (long long)BHN * TT * HDP) return;
    int d  = (int)(i % HDP);
    long long r = i / HDP;
    int t  = (int)(r % TT);
    int bh = (int)(r / TT);
    int b  = bh / HH, h = bh % HH;
    long long row = ((long long)b * TT + t) * C3P;
    float q = 0.f, k = 0.f, v = 0.f;
    if (d < HD) {
        int col = h * HD + d;
        q = QKV[row + col] * 0.13608276348795434f;   // 1/sqrt(54)
        k = QKV[row + CC + col];
        v = QKV[row + 2 * CC + col];
    }
    Qh[((long long)bh * TT + t) * HDP + d] = (_Float16)q;
    Kt[((long long)bh * HDP + d) * TT + t] = (_Float16)k;   // B-operand: K rows = head-dim
    Vh[((long long)bh * TT + t) * HDP + d] = (_Float16)v;   // B-operand: K rows = keys
}

// ---------------- causal softmax row -> fp16 probabilities ----------------
__global__ void k_softmax(const float* __restrict__ S, _Float16* __restrict__ P) {
    int q = blockIdx.x, z = blockIdx.y;
    const float* row = S + ((long long)z * TT + q) * TT;
    _Float16* prow   = P + ((long long)z * TT + q) * TT;
    __shared__ float sh[256];
    int nv = q + 1;
    float mx = -1e30f;
    for (int j = threadIdx.x; j < nv; j += 256) mx = fmaxf(mx, row[j]);
    sh[threadIdx.x] = mx; __syncthreads();
    for (int o = 128; o > 0; o >>= 1) { if (threadIdx.x < o) sh[threadIdx.x] = fmaxf(sh[threadIdx.x], sh[threadIdx.x + o]); __syncthreads(); }
    mx = sh[0]; __syncthreads();
    float sum = 0.f;
    for (int j = threadIdx.x; j < nv; j += 256) sum += __expf(row[j] - mx);
    sh[threadIdx.x] = sum; __syncthreads();
    for (int o = 128; o > 0; o >>= 1) { if (threadIdx.x < o) sh[threadIdx.x] += sh[threadIdx.x + o]; __syncthreads(); }
    float inv = 1.f / sh[0];
    for (int j = threadIdx.x; j < TT; j += 256)
        prow[j] = (_Float16)((j < nv) ? __expf(row[j] - mx) * inv : 0.f);
}

// ---------------- merge heads: [BH][T][HDP] fp32 -> [BT][CP] fp16 ----------------
__global__ void k_merge(const float* __restrict__ Y, _Float16* __restrict__ Ym) {
    long long i = (long long)blockIdx.x * 256 + threadIdx.x;
    if (i >= (long long)BT * CP) return;
    int c = (int)(i % CP);
    long long row = i / CP;
    float v = 0.f;
    if (c < CC) {
        int h = c / HD, d = c % HD;
        int b = (int)(row / TT), t = (int)(row % TT);
        v = Y[(((long long)(b * HH + h) * TT) + t) * HDP + d];
    }
    Ym[i] = (_Float16)v;
}

// ---------------- loss ----------------
__global__ void k_zero2(float* a) { if (threadIdx.x < 2) a[threadIdx.x] = 0.f; }

__global__ void k_loss_row(const float* __restrict__ logits, const int* __restrict__ tgt,
                           float* __restrict__ accum) {
    int row = blockIdx.x;
    const float* lr = logits + (long long)row * VV;
    __shared__ float sh[256];
    float mx = -1e30f;
    for (int j = threadIdx.x; j < VV; j += 256) mx = fmaxf(mx, lr[j]);
    sh[threadIdx.x] = mx; __syncthreads();
    for (int o = 128; o > 0; o >>= 1) { if (threadIdx.x < o) sh[threadIdx.x] = fmaxf(sh[threadIdx.x], sh[threadIdx.x + o]); __syncthreads(); }
    mx = sh[0]; __syncthreads();
    float sum = 0.f;
    for (int j = threadIdx.x; j < VV; j += 256) sum += __expf(lr[j] - mx);
    sh[threadIdx.x] = sum; __syncthreads();
    for (int o = 128; o > 0; o >>= 1) { if (threadIdx.x < o) sh[threadIdx.x] += sh[threadIdx.x + o]; __syncthreads(); }
    if (threadIdx.x == 0) {
        int t = tgt[row];
        if (t != -1) {
            int tc = t < 0 ? 0 : (t >= VV ? VV - 1 : t);
            float logp = lr[tc] - mx - logf(sh[0]);
            atomicAdd(&accum[0], -logp);
            atomicAdd(&accum[1], 1.f);
        }
    }
}

__global__ void k_loss_final(const float* __restrict__ accum, float* __restrict__ out) {
    if (threadIdx.x == 0) out[0] = accum[0] / fmaxf(accum[1], 1.f);
}

// ---------------- host side ----------------
static inline void launch_gemm(hipStream_t s,
                               const _Float16* A, long long sA, int lda,
                               const _Float16* B, long long sB, int ldb,
                               const float* bias,
                               float* C, long long sC, int ldc,
                               _Float16* Ch, long long sCh, int ldch,
                               int M, int Ntiles, int K, int Nvalid, int Nstore,
                               int Z, int flags) {
    dim3 g(Ntiles, M / 64, Z), b(32);
    k_gemm<<<g, b, 0, s>>>(A, sA, lda, B, sB, ldb, bias, C, sC, ldc,
                           Ch, sCh, ldch, K, Nvalid, Nstore, flags);
}

extern "C" void kernel_launch(void* const* d_in, const int* in_sizes, int n_in,
                              void* d_out, int out_size, void* d_ws, size_t ws_size,
                              hipStream_t stream) {
    (void)in_sizes; (void)n_in; (void)out_size; (void)ws_size;
    const int*   idx     = (const int*)d_in[0];
    const int*   targets = (const int*)d_in[1];
    const float* wte     = (const float*)d_in[2];
    const float* wpe     = (const float*)d_in[3];
    const float* ln1_w   = (const float*)d_in[4];
    const float* ln1_b   = (const float*)d_in[5];
    const float* attn_w  = (const float*)d_in[6];
    const float* attn_b  = (const float*)d_in[7];
    const float* proj_w  = (const float*)d_in[8];
    const float* proj_b  = (const float*)d_in[9];
    const float* ln2_w   = (const float*)d_in[10];
    const float* ln2_b   = (const float*)d_in[11];
    const float* fc_w    = (const float*)d_in[12];
    const float* fc_b    = (const float*)d_in[13];
    const float* fcp_w   = (const float*)d_in[14];
    const float* fcp_b   = (const float*)d_in[15];
    const float* lnf_w   = (const float*)d_in[16];
    const float* lnf_b   = (const float*)d_in[17];

    // bump allocator over d_ws (~350 MB total)
    char* wp = (char*)d_ws;
    auto alloc = [&](size_t bytes) -> void* {
        uintptr_t p = (uintptr_t)wp;
        p = (p + 255) & ~(uintptr_t)255;
        wp = (char*)(p + bytes);
        return (void*)p;
    };
    float*     X    = (float*)    alloc((size_t)BT * CP * 4);
    _Float16*  Hh   = (_Float16*) alloc((size_t)BT * CP * 2);
    float*     QKV  = (float*)    alloc((size_t)BT * C3P * 4);
    _Float16*  Qh   = (_Float16*) alloc((size_t)BHN * TT * HDP * 2);
    _Float16*  Kt   = (_Float16*) alloc((size_t)BHN * TT * HDP * 2);
    _Float16*  Vh   = (_Float16*) alloc((size_t)BHN * TT * HDP * 2);
    float*     S    = (float*)    alloc((size_t)BHN * TT * TT * 4);
    _Float16*  P    = (_Float16*) alloc((size_t)BHN * TT * TT * 2);
    float*     Ybh  = (float*)    alloc((size_t)BHN * TT * HDP * 4);
    _Float16*  Ym   = (_Float16*) alloc((size_t)BT * CP * 2);
    _Float16*  Midh = (_Float16*) alloc((size_t)BT * C4P * 2);
    _Float16*  Wq   = (_Float16*) alloc((size_t)CP * C3P * 2);
    _Float16*  Wp   = (_Float16*) alloc((size_t)CP * CP * 2);
    _Float16*  Wf   = (_Float16*) alloc((size_t)CP * C4P * 2);
    _Float16*  Wf2  = (_Float16*) alloc((size_t)C4P * CP * 2);
    _Float16*  WteT = (_Float16*) alloc((size_t)CP * VP * 2);
    float*     acc  = (float*)    alloc(256);
    float*     logits = (float*)d_out;
    float*     loss   = logits + (long long)BT * VV;

    dim3 b256(256);
    auto nblk = [](long long total) { return dim3((unsigned)((total + 255) / 256)); };

    // embedding
    k_embed<<<nblk((long long)BT * CP), b256, 0, stream>>>(idx, wte, wpe, X);
    // lm-head weight (transposed, fp16) — done once
    k_wteT<<<nblk((long long)CP * VP), b256, 0, stream>>>(wte, WteT);

    for (int l = 0; l < LL; ++l) {
        // per-layer weight conversion (fp32 -> padded fp16); L2-resident working set
        k_cvt_pad<<<nblk((long long)CP * C3P), b256, 0, stream>>>(attn_w + (long long)l * CC * C3, Wq, CC, C3, C3P, (long long)CP * C3P);
        k_cvt_pad<<<nblk((long long)CP * CP),  b256, 0, stream>>>(proj_w + (long long)l * CC * CC, Wp, CC, CC, CP, (long long)CP * CP);
        k_cvt_pad<<<nblk((long long)CP * C4P), b256, 0, stream>>>(fc_w   + (long long)l * CC * C4, Wf, CC, C4, C4P, (long long)CP * C4P);
        k_cvt_pad<<<nblk((long long)C4P * CP), b256, 0, stream>>>(fcp_w  + (long long)l * C4 * CC, Wf2, C4, CC, CP, (long long)C4P * CP);

        // LN1 -> fp16
        k_ln<<<dim3(BT), b256, 0, stream>>>(X, ln1_w + l * CC, ln1_b + l * CC, Hh);
        // qkv = Hh @ Wq + b
        launch_gemm(stream, Hh, 0, CP, Wq, 0, C3P, attn_b + (long long)l * C3,
                    QKV, 0, C3P, nullptr, 0, 0, BT, C3P / 16, CP, C3, C3P, 1, 0);
        // split heads (scale folded into Q)
        k_split<<<nblk((long long)BHN * TT * HDP), b256, 0, stream>>>(QKV, Qh, Kt, Vh);
        // scores = Q @ K^T  (batched over 80 heads)
        launch_gemm(stream, Qh, (long long)TT * HDP, HDP, Kt, (long long)HDP * TT, TT, nullptr,
                    S, (long long)TT * TT, TT, nullptr, 0, 0, TT, TT / 16, HDP, TT, TT, BHN, 0);
        // causal softmax -> fp16 probs
        k_softmax<<<dim3(TT, BHN), b256, 0, stream>>>(S, P);
        // y = P @ V
        launch_gemm(stream, P, (long long)TT * TT, TT, Vh, (long long)TT * HDP, HDP, nullptr,
                    Ybh, (long long)TT * HDP, HDP, nullptr, 0, 0, TT, HDP / 16, TT, HDP, HDP, BHN, 0);
        // merge heads -> fp16
        k_merge<<<nblk((long long)BT * CP), b256, 0, stream>>>(Ybh, Ym);
        // x += y @ Wp + b   (residual)
        launch_gemm(stream, Ym, 0, CP, Wp, 0, CP, proj_b + (long long)l * CC,
                    X, 0, CP, nullptr, 0, 0, BT, CP / 16, CP, CC, CP, 1, 2);
        // LN2 -> fp16
        k_ln<<<dim3(BT), b256, 0, stream>>>(X, ln2_w + l * CC, ln2_b + l * CC, Hh);
        // mid = gelu(Hh @ Wf + b)  (fp16 out only)
        launch_gemm(stream, Hh, 0, CP, Wf, 0, C4P, fc_b + (long long)l * C4,
                    nullptr, 0, 0, Midh, 0, C4P, BT, C4P / 16, CP, C4, C4P, 1, 1);
        // x += mid @ Wf2 + b  (residual)
        launch_gemm(stream, Midh, 0, C4P, Wf2, 0, CP, fcp_b + (long long)l * CC,
                    X, 0, CP, nullptr, 0, 0, BT, CP / 16, C4P, CC, CP, 1, 2);
    }

    // final LN -> fp16
    k_ln<<<dim3(BT), b256, 0, stream>>>(X, lnf_w, lnf_b, Hh);
    // logits = Hh @ wte^T   (edge-masked store into unpadded [BT][V])
    launch_gemm(stream, Hh, 0, CP, WteT, 0, VP, nullptr,
                logits, 0, VV, nullptr, 0, 0, BT, VP / 16, CP, VV, VV, 1, 0);

    // loss
    k_zero2<<<dim3(1), dim3(32), 0, stream>>>(acc);
    k_loss_row<<<dim3(BT), b256, 0, stream>>>(logits, targets, acc);
    k_loss_final<<<dim3(1), dim3(32), 0, stream>>>(acc, loss);
}
